// NeuralMemory_55886114455593
// MI455X (gfx1250) — matrix operand order repaired and checked by reading, not compiled, not run on hardware
//
#include <hip/hip_runtime.h>
#include <hip/hip_bf16.h>
#include <cstdint>
#include <cstddef>

// ---------------------------------------------------------------------------
// Problem constants
// ---------------------------------------------------------------------------
constexpr int DIM   = 512;
constexpr int CH    = 64;     // chunk length
constexpr int BATCH = 16;
constexpr int SEQ   = 8192;
constexpr int NC    = SEQ / CH;   // 128 chunks (sequential recurrence)

constexpr float ETAV   = 0.9f;
constexpr float ALPHAV = 0.1f;
constexpr float THETAV = 0.1f;
constexpr float WDV    = 1e-4f;
constexpr float EPSV   = 1e-12f;

// ---------------------------------------------------------------------------
// Types for CDNA5 WMMA (wave32) + TDM descriptors
// ---------------------------------------------------------------------------
typedef __bf16 bf16_t;
typedef __attribute__((ext_vector_type(16))) __bf16 v16bf;
typedef __attribute__((ext_vector_type(8)))  __bf16 v8bf;
typedef __attribute__((ext_vector_type(8)))  float  v8f;
typedef __attribute__((ext_vector_type(4)))  float  v4f;
typedef __attribute__((ext_vector_type(4)))  unsigned int u32x4;
typedef __attribute__((ext_vector_type(8)))  int i32x8;
typedef __attribute__((ext_vector_type(4)))  int i32x4;

__device__ __forceinline__ bf16_t f2bf(float f) { return (bf16_t)f; }
__device__ __forceinline__ float  bf2f(bf16_t h) { return (float)h; }

// ---------------------------------------------------------------------------
// Workspace layout (bf16 weights ~6.8MB, L2-resident; fp32 state/staging)
//   *T = [out][in]  -> B-fragments for forward GEMMs
//   *p = [in][out]  -> B-fragments for backward GEMMs (B = W^T => row-major W)
// ---------------------------------------------------------------------------
constexpr size_t oWkT = 0;
constexpr size_t oWvT = oWkT + (size_t)DIM * DIM;
constexpr size_t oWqT = oWvT + (size_t)DIM * DIM;
constexpr size_t oWoT = oWqT + (size_t)DIM * DIM;
constexpr size_t oWgT = oWoT + (size_t)DIM * DIM;
constexpr size_t oW2T = oWgT + (size_t)DIM * DIM;
constexpr size_t oW3T = oW2T + (size_t)DIM * DIM;
constexpr size_t oW1T = oW3T + (size_t)DIM * DIM;          // [512][1024]
constexpr size_t oW1p = oW1T + (size_t)DIM * 2 * DIM;      // [1024][512]
constexpr size_t oW2p = oW1p + (size_t)2 * DIM * DIM;
constexpr size_t oW3p = oW2p + (size_t)DIM * DIM;
constexpr size_t BF_ELEMS = oW3p + (size_t)DIM * DIM;
constexpr size_t BF_BYTES = BF_ELEMS * 2;                  // 256B aligned

constexpr size_t fMem  = 0;                                   // [16][512]
constexpr size_t fMom  = fMem  + (size_t)BATCH * DIM;
constexpr size_t fZ1   = fMom  + (size_t)BATCH * DIM;         // [16][64][512]
constexpr size_t fZ2   = fZ1   + (size_t)BATCH * CH * DIM;
constexpr size_t fPred = fZ2   + (size_t)BATCH * CH * DIM;
constexpr size_t fV    = fPred + (size_t)BATCH * CH * DIM;

// Dynamic LDS partition (needs CDNA5's 320KB WGP LDS):
//   sIn [64][1024] bf16 : x fp32 staging (TDM dest) -> concat(k|q, mem)  (128 KB)
//   sA  [64][512]  bf16 : x / h1 / dpred / dh1 / ret                     ( 64 KB)
//   sQ  [64][512]  bf16 : q stash until retrieval                        ( 64 KB)
constexpr unsigned S_A_OFF   = 64u * 1024u * 2u;               // 131072
constexpr unsigned S_Q_OFF   = S_A_OFF + 64u * 512u * 2u;      // 196608
constexpr unsigned S_F_OFF   = S_Q_OFF + 64u * 512u * 2u;      // 262144
constexpr unsigned SMEM_BYTES = S_F_OFF + (512u + 512u + 1024u + 512u) * 4u; // 272384

// ---------------------------------------------------------------------------
// WMMA fragment loads per CDNA5 ISA 7.12.2 layouts (wave32).
// A 16x32 bf16: lanes 0-15 row M, K {k0..k0+7, k0+16..k0+23};
//               lanes 16-31 row M, K {k0+8..k0+15, k0+24..k0+31}.
// ---------------------------------------------------------------------------
__device__ __forceinline__ v16bf load_a_frag(const bf16_t* __restrict__ A,
                                             int lda, int m0, int k0, int lane) {
  const int row  = m0 + (lane & 15);
  const int koff = k0 + ((lane >> 4) << 3);    // +0 or +8
  const bf16_t* p = A + (size_t)row * lda + koff;
  v8bf lo = *(const v8bf*)p;
  v8bf hi = *(const v8bf*)(p + 16);
  return __builtin_shufflevector(lo, hi, 0,1,2,3,4,5,6,7,8,9,10,11,12,13,14,15);
}

// B 32x16 bf16 from Bt stored [N][K]:
// lanes 0-15: col n0+lane, K {k0..k0+7, k0+16..k0+23}; lanes 16-31: col n0+lane-16, K {k0+8..k0+15, k0+24..k0+31}.
__device__ __forceinline__ v16bf load_b_frag(const bf16_t* __restrict__ Bt,
                                             int ldb, int n0, int k0, int lane) {
  const int col  = n0 + (lane & 15);
  const int koff = k0 + ((lane >> 4) << 3);    // +0 or +8
  const bf16_t* p = Bt + (size_t)col * ldb + koff;
  v8bf lo = *(const v8bf*)p;
  v8bf hi = *(const v8bf*)(p + 16);
  return __builtin_shufflevector(lo, hi, 0,1,2,3,4,5,6,7,8,9,10,11,12,13,14,15);
}

#define WMMA_BF16(a, b, c) \
  __builtin_amdgcn_wmma_f32_16x16x32_bf16(false, (a), false, (b), (short)0, (c), false, false)

// C = A[64,K] x B[K,512] (+bias). 8 waves; each wave owns two 32-col groups.
// Within a group, one set of A-fragments feeds TWO n-tiles (halves LDS reads
// per WMMA: 1 ds_load_b128 per v_wmma instead of 2). 8 f32 accumulators.
template <typename Epi>
__device__ __forceinline__ void gemm64x512(const bf16_t* __restrict__ A, int K,
                                           const bf16_t* __restrict__ Bt,
                                           const float* __restrict__ bias,
                                           Epi epi) {
  const int lane = threadIdx.x & 31;
  const int wave = threadIdx.x >> 5;
  for (int np = wave; np < 16; np += 8) {
    const int n0 = np << 5;                    // 32-column group
    v8f a00 = {}, a01 = {}, a02 = {}, a03 = {};   // n-tile 0, m-tiles 0..3
    v8f a10 = {}, a11 = {}, a12 = {}, a13 = {};   // n-tile 1, m-tiles 0..3
    for (int k0 = 0; k0 < K; k0 += 32) {
      if (k0 + 64 <= K) {   // stream next weight columns (global_prefetch_b8)
        __builtin_prefetch((const void*)(Bt + (size_t)(n0 + (lane & 15)) * K + k0 + 64), 0, 1);
        __builtin_prefetch((const void*)(Bt + (size_t)(n0 + 16 + (lane & 15)) * K + k0 + 64), 0, 1);
      }
      const v16bf b0 = load_b_frag(Bt, K, n0,      k0, lane);
      const v16bf b1 = load_b_frag(Bt, K, n0 + 16, k0, lane);
      const v16bf f0 = load_a_frag(A, K, 0,  k0, lane);
      const v16bf f1 = load_a_frag(A, K, 16, k0, lane);
      const v16bf f2 = load_a_frag(A, K, 32, k0, lane);
      const v16bf f3 = load_a_frag(A, K, 48, k0, lane);
      a00 = WMMA_BF16(f0, b0, a00);  a10 = WMMA_BF16(f0, b1, a10);
      a01 = WMMA_BF16(f1, b0, a01);  a11 = WMMA_BF16(f1, b1, a11);
      a02 = WMMA_BF16(f2, b0, a02);  a12 = WMMA_BF16(f2, b1, a12);
      a03 = WMMA_BF16(f3, b0, a03);  a13 = WMMA_BF16(f3, b1, a13);
    }
    const int rb   = (lane >> 4) << 3;          // lanes>=16 hold M = 8..15 of tile
    const int colA = n0 + (lane & 15);
    const int colB = colA + 16;
    const float bbA = bias ? bias[colA] : 0.0f;
    const float bbB = bias ? bias[colB] : 0.0f;
#pragma unroll
    for (int r = 0; r < 8; ++r) {
      epi(rb + r,      colA, a00[r] + bbA);
      epi(16 + rb + r, colA, a01[r] + bbA);
      epi(32 + rb + r, colA, a02[r] + bbA);
      epi(48 + rb + r, colA, a03[r] + bbA);
      epi(rb + r,      colB, a10[r] + bbB);
      epi(16 + rb + r, colB, a11[r] + bbB);
      epi(32 + rb + r, colB, a12[r] + bbB);
      epi(48 + rb + r, colB, a13[r] + bbB);
    }
  }
}

// ---------------------------------------------------------------------------
// TDM: stage one contiguous 64x512 fp32 chunk (128KB) global -> LDS.
// D# per CDNA5 ISA ch.8: 1-D tile, data_size=4B, tile_dim0=32768 elements.
// Issued by wave 0 only; completion via s_wait_tensorcnt + barrier.
// ---------------------------------------------------------------------------
__device__ __forceinline__ void tdm_load_chunk_to_lds(const float* gsrc, void* ldsdst) {
  const unsigned long long ga = (unsigned long long)(uintptr_t)gsrc;
  const unsigned int lds_addr = (unsigned int)(uintptr_t)ldsdst;  // low 32b = LDS byte addr
  u32x4 g0;
  g0[0] = 1u;                                             // count=1, user mode
  g0[1] = lds_addr;                                       // lds_addr [63:32]
  g0[2] = (unsigned int)(ga & 0xffffffffu);               // global_addr low
  g0[3] = (unsigned int)((ga >> 32) & 0x01ffffffu) | (2u << 30);  // ga[56:32] | type=2
  i32x8 g1;
  g1[0] = (int)(2u << 16);          // workgroup_mask=0, data_size=2 (4 bytes)
  g1[1] = (int)0x80000000;          // tensor_dim0[15:0]=32768 -> word1[31:16]
  g1[2] = (int)(1u << 16);          // tensor_dim0 hi=0 | tensor_dim1[15:0]=1
  g1[3] = (int)0x80000000;          // tensor_dim1 hi=0 | tile_dim0=32768
  g1[4] = 1;                        // tile_dim1=1, tile_dim2=0
  g1[5] = 32768;                    // tensor_dim0_stride low 32
  g1[6] = 0;                        // stride hi | tensor_dim1_stride low
  g1[7] = 0;
  i32x4 gz = {0, 0, 0, 0};          // groups 2/3 unused (<=2D tensor)
#if __clang_major__ >= 23
  i32x8 gz8 = {0, 0, 0, 0, 0, 0, 0, 0};
  __builtin_amdgcn_tensor_load_to_lds(g0, g1, gz, gz, gz8, 0);
#else
  __builtin_amdgcn_tensor_load_to_lds(g0, g1, gz, gz, 0);
#endif
  __builtin_amdgcn_s_wait_tensorcnt(0);
}

// ---------------------------------------------------------------------------
// Prep kernels (weights fp32 -> bf16, transposed + plain; state zeroing)
// ---------------------------------------------------------------------------
__global__ void k_transpose_bf16(const float* __restrict__ W, bf16_t* __restrict__ WT,
                                 int rows, int cols) {
  int i = blockIdx.x * blockDim.x + threadIdx.x;
  if (i >= rows * cols) return;
  int r = i / cols, c = i % cols;
  WT[(size_t)c * rows + r] = f2bf(W[i]);
}

__global__ void k_copy_bf16(const float* __restrict__ W, bf16_t* __restrict__ Wp, int n) {
  int i = blockIdx.x * blockDim.x + threadIdx.x;
  if (i < n) Wp[i] = f2bf(W[i]);
}

__global__ void k_zero_f32(float* __restrict__ p, int n) {
  int i = blockIdx.x * blockDim.x + threadIdx.x;
  if (i < n) p[i] = 0.0f;
}

// ---------------------------------------------------------------------------
// One recurrence step: grid = 16 workgroups (one batch element each),
// 256 threads = 8 wave32. 13 WMMA GEMMs per workgroup.
// ---------------------------------------------------------------------------
__global__ __launch_bounds__(256) void chunk_step_kernel(
    int c, const float* __restrict__ x,
    const float* __restrict__ bk, const float* __restrict__ bv,
    const float* __restrict__ bq, const float* __restrict__ bo,
    const float* __restrict__ b1, const float* __restrict__ b2,
    const float* __restrict__ b3, const float* __restrict__ bg,
    const bf16_t* __restrict__ wbf, float* __restrict__ fws,
    float* __restrict__ out) {
  extern __shared__ unsigned char smem[];
  bf16_t* sIn  = (bf16_t*)(smem);              // [64][1024]
  bf16_t* sA   = (bf16_t*)(smem + S_A_OFF);    // [64][512]
  bf16_t* sQ   = (bf16_t*)(smem + S_Q_OFF);    // [64][512]
  float*  sF   = (float*)(smem + S_F_OFF);
  float*  sCol  = sF;          // [512] col reduce: k-mean, then grad sum
  float*  sGate = sF + 512;    // [512]
  float*  sRP   = sF + 1024;   // [64][4][4] row partials
  float*  sStat = sF + 2048;   // [64][8]: S, |p|, |v|, cos, w, gscale
  bf16_t* sH2 = sIn;           // [64][512] alias of first 64KB of sIn (stage-safe)

  const int b   = blockIdx.x;
  const int tid = threadIdx.x;

  const float* xc = x + ((size_t)b * SEQ + (size_t)c * CH) * DIM;
  float* z1   = fws + fZ1   + (size_t)b * CH * DIM;
  float* z2   = fws + fZ2   + (size_t)b * CH * DIM;
  float* pred = fws + fPred + (size_t)b * CH * DIM;
  float* vbuf = fws + fV    + (size_t)b * CH * DIM;
  float* memv = fws + fMem + (size_t)b * DIM;
  float* momv = fws + fMom + (size_t)b * DIM;

  // ---- TDM: async stage x chunk (fp32) into sIn region ----
  if ((tid >> 5) == 0) tdm_load_chunk_to_lds(xc, (void*)sIn);
  for (int i = tid; i < DIM; i += 256) sCol[i] = 0.0f;
  __syncthreads();

  // ---- convert staged x to bf16 A-matrix ----
  {
    const float* xstage = (const float*)sIn;
    for (int i = tid * 4; i < CH * DIM; i += 256 * 4) {
      v4f xv = *(const v4f*)(xstage + i);
      sA[i + 0] = f2bf(xv.x); sA[i + 1] = f2bf(xv.y);
      sA[i + 2] = f2bf(xv.z); sA[i + 3] = f2bf(xv.w);
    }
  }
  __syncthreads();

  // ---- projections: k -> sIn left half (+ column sums for gate), v -> fp32, q -> stash ----
  gemm64x512(sA, DIM, wbf + oWkT, bk, [&](int r, int cc, float v) {
    sIn[(size_t)r * 1024 + cc] = f2bf(v);
    atomicAdd(&sCol[cc], v);                      // ds_add_f32
  });
  gemm64x512(sA, DIM, wbf + oWvT, bv, [&](int r, int cc, float v) {
    vbuf[(size_t)r * DIM + cc] = v;
  });
  gemm64x512(sA, DIM, wbf + oWqT, bq, [&](int r, int cc, float v) {
    sQ[(size_t)r * DIM + cc] = f2bf(v);
  });
  // broadcast current memory into right half of concat input
  for (int d = tid; d < DIM; d += 256) {
    bf16_t mb = f2bf(memv[d]);
    for (int r = 0; r < CH; ++r) sIn[(size_t)r * 1024 + DIM + d] = mb;
  }
  __syncthreads();

  // ---- momentum gate: sigmoid(mean_k @ Wg + bg) (small matvec, VALU) ----
  for (int d = tid; d < DIM; d += 256) {
    const bf16_t* wr = wbf + oWgT + (size_t)d * DIM;
    float acc = bg[d];
    for (int k = 0; k < DIM; ++k) acc += (sCol[k] * (1.0f / 64.0f)) * bf2f(wr[k]);
    sGate[d] = 1.0f / (1.0f + __expf(-acc));
  }

  // ---- loss-branch MLP forward (keep pre-activations for silu') ----
  gemm64x512(sIn, 2 * DIM, wbf + oW1T, b1, [&](int r, int cc, float z) {
    z1[(size_t)r * DIM + cc] = z;
    sA[(size_t)r * DIM + cc] = f2bf(z / (1.0f + __expf(-z)));
  });
  __syncthreads();
  gemm64x512(sA, DIM, wbf + oW2T, b2, [&](int r, int cc, float z) {
    z2[(size_t)r * DIM + cc] = z;
    sH2[(size_t)r * DIM + cc] = f2bf(z / (1.0f + __expf(-z)));
  });
  __syncthreads();
  for (int i = tid; i < DIM; i += 256) sCol[i] = 0.0f;  // reuse for grad sum
  gemm64x512(sH2, DIM, wbf + oW3T, b3, [&](int r, int cc, float z) {
    pred[(size_t)r * DIM + cc] = z;
  });
  __syncthreads();

  // ---- per-row loss statistics (4 threads per row) ----
  {
    const int r = tid >> 2, sub = tid & 3;
    const float* pr = pred + (size_t)r * DIM;
    const float* vr = vbuf + (size_t)r * DIM;
    float mse = 0, p2 = 0, v2 = 0, pv = 0;
    for (int d = sub * 128; d < sub * 128 + 128; ++d) {
      float p = pr[d], vv = vr[d], df = p - vv;
      mse += df * df; p2 += p * p; v2 += vv * vv; pv += p * vv;
    }
    float* o = sRP + (size_t)(r * 4 + sub) * 4;
    o[0] = mse; o[1] = p2; o[2] = v2; o[3] = pv;
  }
  __syncthreads();
  if (tid < CH) {
    const int r = tid;
    float mse = 0, p2 = 0, v2 = 0, pv = 0;
    for (int s = 0; s < 4; ++s) {
      const float* o = sRP + (size_t)(r * 4 + s) * 4;
      mse += o[0]; p2 += o[1]; v2 += o[2]; pv += o[3];
    }
    const float pn = sqrtf(p2), vn = sqrtf(v2);
    const float cosv = pv / ((pn + EPSV) * (vn + EPSV));
    const float S = mse + 512.0f * 0.5f * (1.0f - cosv);   // sum_d loss[r,d]
    sStat[r * 8 + 0] = S; sStat[r * 8 + 1] = pn;
    sStat[r * 8 + 2] = vn; sStat[r * 8 + 3] = cosv;
  }
  __syncthreads();
  if (tid == 0) {
    // softmax over chunk rows of s = S/512, exact grad through softmax:
    // dL/dloss[r,d] = w_r * (1 + (S_r - sum_r' w_r' S_r') / 512)
    float mx = -3.4e38f;
    for (int r = 0; r < CH; ++r) mx = fmaxf(mx, sStat[r * 8] * (1.0f / 512.0f));
    float sum = 0.0f;
    for (int r = 0; r < CH; ++r) {
      float e = __expf(sStat[r * 8] * (1.0f / 512.0f) - mx);
      sStat[r * 8 + 4] = e; sum += e;
    }
    float Sbar = 0.0f;
    for (int r = 0; r < CH; ++r) {
      float w = sStat[r * 8 + 4] / sum;
      sStat[r * 8 + 4] = w; Sbar += w * sStat[r * 8];
    }
    for (int r = 0; r < CH; ++r)
      sStat[r * 8 + 5] = sStat[r * 8 + 4] *
                         (1.0f + (sStat[r * 8] - Sbar) * (1.0f / 512.0f));
  }
  __syncthreads();

  // ---- dL/dpred (mse + exact cosine-distance derivative) -> bf16 A matrix ----
  {
    const int r = tid >> 2, sub = tid & 3;
    const float gs = sStat[r * 8 + 5], pn = sStat[r * 8 + 1];
    const float vn = sStat[r * 8 + 2], cosv = sStat[r * 8 + 3];
    const float ipn = 1.0f / (pn + EPSV), ivn = 1.0f / (vn + EPSV);
    const float* pr = pred + (size_t)r * DIM;
    const float* vr = vbuf + (size_t)r * DIM;
    for (int d = sub * 128; d < sub * 128 + 128; ++d) {
      float p = pr[d], vv = vr[d];
      float dp = gs * (2.0f * (p - vv) - 256.0f * (vv * ivn - cosv * p * ipn) * ipn);
      sA[(size_t)r * DIM + d] = f2bf(dp);
    }
  }
  __syncthreads();

  // ---- backward through MLP to memory gradient ----
  gemm64x512(sA, DIM, wbf + oW3p, nullptr, [&](int r, int cc, float g) {
    float z = z2[(size_t)r * DIM + cc], s = 1.0f / (1.0f + __expf(-z));
    sH2[(size_t)r * DIM + cc] = f2bf(g * s * (1.0f + z * (1.0f - s)));
  });
  __syncthreads();
  gemm64x512(sH2, DIM, wbf + oW2p, nullptr, [&](int r, int cc, float g) {
    float z = z1[(size_t)r * DIM + cc], s = 1.0f / (1.0f + __expf(-z));
    sA[(size_t)r * DIM + cc] = f2bf(g * s * (1.0f + z * (1.0f - s)));
  });
  __syncthreads();
  // grad wrt mem = sum over chunk rows of (dh1 @ W1^T)[:, 512:1024]
  gemm64x512(sA, DIM, wbf + oW1p + (size_t)DIM * DIM, nullptr,
             [&](int r, int cc, float g) { atomicAdd(&sCol[cc], g); });
  __syncthreads();

  // ---- gated momentum state update + rebuild retrieval input ----
  for (int d = tid; d < DIM; d += 256) {
    float g  = sCol[d];
    float mo = sGate[d] * (ETAV * momv[d] - THETAV * g);
    float me = (1.0f - ALPHAV - WDV) * memv[d] + mo;
    momv[d] = mo; memv[d] = me;
    bf16_t mb = f2bf(me);
    for (int r = 0; r < CH; ++r) sIn[(size_t)r * 1024 + DIM + d] = mb;
  }
  for (int i = tid; i < CH * DIM; i += 256)
    sIn[(size_t)(i >> 9) * 1024 + (i & 511)] = sQ[i];
  __syncthreads();

  // ---- retrieval MLP + output projection ----
  gemm64x512(sIn, 2 * DIM, wbf + oW1T, b1, [&](int r, int cc, float z) {
    sA[(size_t)r * DIM + cc] = f2bf(z / (1.0f + __expf(-z)));
  });
  __syncthreads();
  gemm64x512(sA, DIM, wbf + oW2T, b2, [&](int r, int cc, float z) {
    sH2[(size_t)r * DIM + cc] = f2bf(z / (1.0f + __expf(-z)));
  });
  __syncthreads();
  gemm64x512(sH2, DIM, wbf + oW3T, b3, [&](int r, int cc, float z) {
    sA[(size_t)r * DIM + cc] = f2bf(z);
  });
  __syncthreads();
  float* ob = out + ((size_t)b * SEQ + (size_t)c * CH) * DIM;
  gemm64x512(sA, DIM, wbf + oWoT, bo, [&](int r, int cc, float v) {
    ob[(size_t)r * DIM + cc] = v;
  });
}

// ---------------------------------------------------------------------------
// Host entry
// ---------------------------------------------------------------------------
extern "C" void kernel_launch(void* const* d_in, const int* in_sizes, int n_in,
                              void* d_out, int out_size, void* d_ws, size_t ws_size,
                              hipStream_t stream) {
  (void)in_sizes; (void)n_in; (void)out_size; (void)ws_size;
  const float* x  = (const float*)d_in[0];
  const float* Wk = (const float*)d_in[1];
  const float* bk = (const float*)d_in[2];
  const float* Wv = (const float*)d_in[3];
  const float* bv = (const float*)d_in[4];
  const float* Wq = (const float*)d_in[5];
  const float* bq = (const float*)d_in[6];
  const float* Wo = (const float*)d_in[7];
  const float* bo = (const float*)d_in[8];
  const float* W1 = (const float*)d_in[9];
  const float* b1 = (const float*)d_in[10];
  const float* W2 = (const float*)d_in[11];
  const float* b2 = (const float*)d_in[12];
  const float* W3 = (const float*)d_in[13];
  const float* b3 = (const float*)d_in[14];
  const float* Wg = (const float*)d_in[15];
  const float* bg = (const float*)d_in[16];

  bf16_t* wbf = (bf16_t*)d_ws;
  float*  fws = (float*)((char*)d_ws + BF_BYTES);
  float*  out = (float*)d_out;

  const int T = 256;
  auto nb = [](size_t n) { return (unsigned)((n + 255) / 256); };

  // weight conversion (fresh every launch: determinism, ws may be poisoned)
  k_transpose_bf16<<<nb((size_t)DIM * DIM), T, 0, stream>>>(Wk, wbf + oWkT, DIM, DIM);
  k_transpose_bf16<<<nb((size_t)DIM * DIM), T, 0, stream>>>(Wv, wbf + oWvT, DIM, DIM);
  k_transpose_bf16<<<nb((size_t)DIM * DIM), T, 0, stream>>>(Wq, wbf + oWqT, DIM, DIM);
  k_transpose_bf16<<<nb((size_t)DIM * DIM), T, 0, stream>>>(Wo, wbf + oWoT, DIM, DIM);
  k_transpose_bf16<<<nb((size_t)DIM * DIM), T, 0, stream>>>(Wg, wbf + oWgT, DIM, DIM);
  k_transpose_bf16<<<nb((size_t)DIM * DIM), T, 0, stream>>>(W2, wbf + oW2T, DIM, DIM);
  k_transpose_bf16<<<nb((size_t)DIM * DIM), T, 0, stream>>>(W3, wbf + oW3T, DIM, DIM);
  k_transpose_bf16<<<nb((size_t)2 * DIM * DIM), T, 0, stream>>>(W1, wbf + oW1T, 2 * DIM, DIM);
  k_copy_bf16<<<nb((size_t)2 * DIM * DIM), T, 0, stream>>>(W1, wbf + oW1p, 2 * DIM * DIM);
  k_copy_bf16<<<nb((size_t)DIM * DIM), T, 0, stream>>>(W2, wbf + oW2p, DIM * DIM);
  k_copy_bf16<<<nb((size_t)DIM * DIM), T, 0, stream>>>(W3, wbf + oW3p, DIM * DIM);
  // zero mem + mom state
  k_zero_f32<<<nb((size_t)2 * BATCH * DIM), T, 0, stream>>>(fws + fMem, 2 * BATCH * DIM);

  // sequential chunk recurrence (stream order carries the dependence)
  for (int c = 0; c < NC; ++c) {
    chunk_step_kernel<<<dim3(BATCH), dim3(256), SMEM_BYTES, stream>>>(
        c, x, bk, bv, bq, bo, b1, b2, b3, bg, wbf, fws, out);
  }
}
